// TransformerEncoder_10316511445467
// MI455X (gfx1250) — compile-verified, matrix operand order checked by reading, not run on hardware
//
#include <hip/hip_runtime.h>
#include <hip/hip_bf16.h>

// ---------------------------------------------------------------------------
// Types / WMMA helpers (CDNA5 / gfx1250, wave32)
// ---------------------------------------------------------------------------
typedef __attribute__((ext_vector_type(16))) __bf16 v16bf;
typedef __attribute__((ext_vector_type(8)))  float  v8f;

union FragU { v16bf v; unsigned int u[8]; };

__device__ __forceinline__ unsigned short f2bf(float f) {
  unsigned int u = __float_as_uint(f);
  u += 0x7FFFu + ((u >> 16) & 1u);        // round-to-nearest-even
  return (unsigned short)(u >> 16);
}

// Async global->LDS 16-byte copy (CDNA5): VDST = LDS byte offset (low 32 bits
// of the flat shared pointer, per the LDS aperture rule addr[31:0] == offset),
// VADDR = 64-bit global address.  Tracked by ASYNCcnt.
__device__ __forceinline__ void async_g2l_b128(void* lds, const void* gptr) {
  unsigned lo = (unsigned)(unsigned long long)lds;
  asm volatile("global_load_async_to_lds_b128 %0, %1, off"
               :: "v"(lo), "v"(gptr) : "memory");
}
__device__ __forceinline__ void wait_async0() {
  asm volatile("s_wait_asynccnt 0x0" ::: "memory");
}

// 16x32 bf16 fragment (A-layout) from an LDS tile stored (row, k) with k
// contiguous.  B fragments use the same layout because B tiles are staged
// n-major / k-contiguous (weights pre-transposed to [N][K]).
// ISA 7.12.2: lane l -> row = l%16; VGPR v<4: K=(l/16)*8+2v ; v>=4: K=16+...
__device__ __forceinline__ v16bf frag_ld(const unsigned short* tile, int pitch,
                                         int row0, int k0) {
  const int l  = threadIdx.x & 31;
  const int r  = row0 + (l & 15);
  const int h8 = ((l >> 4) & 1) << 3;
  const unsigned int* p = (const unsigned int*)(tile + r * pitch + k0);
  FragU f;
#pragma unroll
  for (int v = 0; v < 8; ++v) {
    const int k = (v < 4) ? (h8 + v * 2) : (16 + h8 + (v - 4) * 2);
    f.u[v] = p[k >> 1];
  }
  return f.v;
}

__device__ __forceinline__ v8f wmma_bf16(v16bf a, v16bf b, v8f c) {
  return __builtin_amdgcn_wmma_f32_16x16x32_bf16(
      false, a, false, b, (short)0, c, false, false);
}

// ---------------------------------------------------------------------------
// Copy kernel (init residual stream x = input)
// ---------------------------------------------------------------------------
__global__ __launch_bounds__(256) void copy_f32(const float* __restrict__ src,
                                                float* __restrict__ dst, int n) {
  int i = blockIdx.x * 256 + threadIdx.x;
  if (i < n) dst[i] = src[i];
}

// ---------------------------------------------------------------------------
// Weight prep: in[K][N] fp32  ->  out[N][K] bf16   (32x32 LDS tile transpose)
// ---------------------------------------------------------------------------
__global__ __launch_bounds__(256) void transpose_cvt(
    const float* __restrict__ in, unsigned short* __restrict__ out,
    int K, int N) {
  __shared__ float t[32][33];
  const int nb = blockIdx.x * 32, kb = blockIdx.y * 32;
  const int tx = threadIdx.x & 31, ty = threadIdx.x >> 5;  // 8 rows per pass
#pragma unroll
  for (int i = 0; i < 32; i += 8)
    t[ty + i][tx] = in[(size_t)(kb + ty + i) * N + nb + tx];
  __syncthreads();
#pragma unroll
  for (int i = 0; i < 32; i += 8)
    out[(size_t)(nb + ty + i) * K + kb + tx] = f2bf(t[tx][ty + i]);
}

// ---------------------------------------------------------------------------
// LayerNorm: one row (D=768) per block; writes f32 (residual) + bf16 (GEMM A)
// ---------------------------------------------------------------------------
__global__ __launch_bounds__(256) void ln_kernel(const float* __restrict__ x,
                                                 const float* __restrict__ g,
                                                 const float* __restrict__ b,
                                                 float* __restrict__ h,
                                                 unsigned short* __restrict__ hb) {
  const int row = blockIdx.x;
  const float* xr = x + (size_t)row * 768;
  __shared__ float red[8];
  const int tid = threadIdx.x;

  float s = 0.f;
  for (int j = tid; j < 768; j += 256) s += xr[j];
  for (int off = 16; off > 0; off >>= 1) s += __shfl_xor(s, off, 32);
  if ((tid & 31) == 0) red[tid >> 5] = s;
  __syncthreads();
  float tot = 0.f;
#pragma unroll
  for (int i = 0; i < 8; ++i) tot += red[i];
  const float mean = tot * (1.0f / 768.0f);
  __syncthreads();

  float vs = 0.f;
  for (int j = tid; j < 768; j += 256) {
    float d = xr[j] - mean;
    vs += d * d;
  }
  for (int off = 16; off > 0; off >>= 1) vs += __shfl_xor(vs, off, 32);
  if ((tid & 31) == 0) red[tid >> 5] = vs;
  __syncthreads();
  float vtot = 0.f;
#pragma unroll
  for (int i = 0; i < 8; ++i) vtot += red[i];
  const float inv = rsqrtf(vtot * (1.0f / 768.0f) + 1e-5f);

  float* hr = h + (size_t)row * 768;
  unsigned short* hbr = hb + (size_t)row * 768;
  for (int j = tid; j < 768; j += 256) {
    float v = (xr[j] - mean) * inv * g[j] + b[j];
    hr[j] = v;
    hbr[j] = f2bf(v);
  }
}

// ---------------------------------------------------------------------------
// bf16 WMMA GEMM:  C[M,N] = A[M,K] @ Bt[N,K]^T  (+bias) (gelu) (+resid)
// BM=128 BN=64 BK=64; 8 waves x (32x32) = 2x2 WMMA accs, 8 WMMAs/stage.
// A and Bt are bf16, k-contiguous -> tiles staged with async b128 copies.
// Output: bf16 (Cb) or f32 (+residual) (Cf).
// ---------------------------------------------------------------------------
#define GBM 128
#define GBN 64
#define GBK 64
#define GP  72   // LDS pitch (bf16 elems): 144-B rows, 16-B aligned

__global__ __launch_bounds__(256) void gemm_bf16(
    const unsigned short* __restrict__ A,   // [M][K] bf16
    const unsigned short* __restrict__ Bt,  // [N][K] bf16
    const float* __restrict__ bias,
    const float* __restrict__ resid,
    float* __restrict__ Cf, unsigned short* __restrict__ Cb,
    int M, int N, int K, int gelu) {
  __shared__ unsigned short As[GBM * GP];  // 18432 B
  __shared__ unsigned short Bs[GBN * GP];  //  9216 B

  const int bm = blockIdx.x * GBM;
  const int bn = blockIdx.y * GBN;
  const int tid = threadIdx.x;
  const int w = tid >> 5;
  const int wm = (w >> 1) * 32;   // 0,32,64,96
  const int wn = (w & 1) * 32;    // 0,32

  v8f acc[2][2] = {};

  for (int k0 = 0; k0 < K; k0 += GBK) {
    __syncthreads();
    // A tile 128x64: 128 rows x 8 b128-chunks = 1024 transfers (4/thread)
#pragma unroll
    for (int t = tid; t < GBM * 8; t += 256) {
      const int r = t >> 3, q = t & 7;
      async_g2l_b128(&As[r * GP + q * 8],
                     &A[(size_t)(bm + r) * K + k0 + q * 8]);
    }
    // B tile 64x64: 64 rows x 8 chunks = 512 transfers (2/thread)
#pragma unroll
    for (int t = tid; t < GBN * 8; t += 256) {
      const int r = t >> 3, q = t & 7;
      async_g2l_b128(&Bs[r * GP + q * 8],
                     &Bt[(size_t)(bn + r) * K + k0 + q * 8]);
    }
    if (k0 + GBK < K)  // global_prefetch_b8 of next A tile
      __builtin_prefetch(&A[(size_t)(bm + (tid >> 1)) * K + k0 + GBK], 0, 1);
    wait_async0();
    __syncthreads();

#pragma unroll
    for (int ks = 0; ks < GBK; ks += 32) {
      v16bf b0 = frag_ld(Bs, GP, wn, ks);
      v16bf b1 = frag_ld(Bs, GP, wn + 16, ks);
#pragma unroll
      for (int i = 0; i < 2; ++i) {
        v16bf a = frag_ld(As, GP, wm + i * 16, ks);
        acc[i][0] = wmma_bf16(a, b0, acc[i][0]);
        acc[i][1] = wmma_bf16(a, b1, acc[i][1]);
      }
    }
  }

  // Epilogue: lane l -> n = l%16, m = vgpr + 8*(l/16)
  const int l = tid & 31;
  const int nl = l & 15;
  const int mh = (l >> 4) * 8;
#pragma unroll
  for (int i = 0; i < 2; ++i) {
#pragma unroll
    for (int j = 0; j < 2; ++j) {
#pragma unroll
      for (int r = 0; r < 8; ++r) {
        const int m = bm + wm + i * 16 + mh + r;
        const int n = bn + wn + j * 16 + nl;
        float v = acc[i][j][r];
        if (bias) v += bias[n];
        if (gelu) v = 0.5f * v * (1.0f + erff(v * 0.70710678f));
        if (Cb) {
          Cb[(size_t)m * N + n] = f2bf(v);
        } else {
          if (resid) v += resid[(size_t)m * N + n];
          Cf[(size_t)m * N + n] = v;
        }
      }
    }
  }
}

// ---------------------------------------------------------------------------
// Attention: one WG = 64 query rows of one (b,h).  qkv is bf16 [tok][2304].
// Full 64x1024 f32 score strip in LDS (256 KB of the 320 KB WGP LDS),
// wave-per-row softmax (in-place f32->bf16 repack), P@V with WMMA.
// Q/K tiles staged with async b128; V staged manually (transposed).
// dyn LDS = 64*1024*4 + 2*64*64*2 = 278528 B.  Output bf16.
// ---------------------------------------------------------------------------
__global__ __launch_bounds__(256) void attn_kernel(
    const unsigned short* __restrict__ qkv, unsigned short* __restrict__ o) {
  extern __shared__ char smem[];
  float*          S  = (float*)smem;                            // [64][1024]
  unsigned short* Qt = (unsigned short*)(smem + 262144);        // [64][64]
  unsigned short* KV = (unsigned short*)(smem + 262144 + 8192); // [64][64]

  const int qb = blockIdx.x;
  const int bh = blockIdx.y;
  const int bb = bh / 12, hh = bh % 12;
  const int tok0 = bb * 1024 + qb * 64;
  const int LDQ = 2304;
  const int qc = hh * 64, kc = 768 + hh * 64, vc = 1536 + hh * 64;
  const int tid = threadIdx.x;
  const int w = tid >> 5, l = tid & 31;
  const int wm = (w >> 1) * 16;
  const int wn = (w & 1) * 32;
  const int nl = l & 15, mh = (l >> 4) * 8;
  const float scale = 0.125f;

  // stage Q block 64x64 bf16: 64 rows x 8 b128-chunks = 512 (2/thread)
#pragma unroll
  for (int t = tid; t < 512; t += 256) {
    const int r = t >> 3, q = t & 7;
    async_g2l_b128(&Qt[r * 64 + q * 8],
                   &qkv[(size_t)(tok0 + r) * LDQ + qc + q * 8]);
  }

  // ---- scores
  for (int kb = 0; kb < 16; ++kb) {
    __syncthreads();
#pragma unroll
    for (int t = tid; t < 512; t += 256) {
      const int r = t >> 3, q = t & 7;   // KV[key][d]
      async_g2l_b128(&KV[r * 64 + q * 8],
                     &qkv[(size_t)(bb * 1024 + kb * 64 + r) * LDQ + kc + q * 8]);
    }
    wait_async0();
    __syncthreads();

    v8f a0 = {}, a1 = {};
#pragma unroll
    for (int ks = 0; ks < 64; ks += 32) {
      v16bf af = frag_ld(Qt, 64, wm, ks);
      v16bf b0 = frag_ld(KV, 64, wn, ks);        // B = K^T: row=key, k=d
      v16bf b1 = frag_ld(KV, 64, wn + 16, ks);
      a0 = wmma_bf16(af, b0, a0);
      a1 = wmma_bf16(af, b1, a1);
    }
#pragma unroll
    for (int r = 0; r < 8; ++r) {
      S[(wm + mh + r) * 1024 + kb * 64 + wn + nl]      = a0[r] * scale;
      S[(wm + mh + r) * 1024 + kb * 64 + wn + 16 + nl] = a1[r] * scale;
    }
  }
  __syncthreads();

  // ---- softmax: wave w owns rows w*8..w*8+7 (1024 f32, 32 per lane)
  for (int rr = 0; rr < 8; ++rr) {
    const int row = w * 8 + rr;
    float* Sr = S + row * 1024;
    float vbuf[32];
    float mx = -3.0e38f;
#pragma unroll
    for (int i = 0; i < 32; ++i) {
      float t = Sr[i * 32 + l];
      vbuf[i] = t;
      mx = fmaxf(mx, t);
    }
    for (int off = 16; off > 0; off >>= 1) mx = fmaxf(mx, __shfl_xor(mx, off, 32));
    float sum = 0.f;
#pragma unroll
    for (int i = 0; i < 32; ++i) {
      float e = __expf(vbuf[i] - mx);
      vbuf[i] = e;
      sum += e;
    }
    for (int off = 16; off > 0; off >>= 1) sum += __shfl_xor(sum, off, 32);
    const float inv = 1.0f / sum;
    unsigned short* Pr = (unsigned short*)Sr;    // in-place bf16 repack
#pragma unroll
    for (int i = 0; i < 32; ++i) Pr[i * 32 + l] = f2bf(vbuf[i] * inv);
  }
  __syncthreads();

  // ---- O = P @ V
  v8f o0 = {}, o1 = {};
  for (int vb = 0; vb < 16; ++vb) {
    __syncthreads();
    for (int e = tid; e < 64 * 64; e += 256) {   // KV[d][key] (transposed V)
      const int key = e >> 6, d = e & 63;
      KV[d * 64 + key] = qkv[(size_t)(bb * 1024 + vb * 64 + key) * LDQ + vc + d];
    }
    __syncthreads();
#pragma unroll
    for (int ks = 0; ks < 64; ks += 32) {
      v16bf af = frag_ld((const unsigned short*)S, 2048, wm, vb * 64 + ks);
      v16bf b0 = frag_ld(KV, 64, wn, ks);        // B = V: row=d, k=key
      v16bf b1 = frag_ld(KV, 64, wn + 16, ks);
      o0 = wmma_bf16(af, b0, o0);
      o1 = wmma_bf16(af, b1, o1);
    }
  }
#pragma unroll
  for (int r = 0; r < 8; ++r) {
    const int m = tok0 + wm + mh + r;
    o[(size_t)m * 768 + hh * 64 + wn + nl]      = f2bf(o0[r]);
    o[(size_t)m * 768 + hh * 64 + wn + 16 + nl] = f2bf(o1[r]);
  }
}

// ---------------------------------------------------------------------------
// Launcher
// ---------------------------------------------------------------------------
extern "C" void kernel_launch(void* const* d_in, const int* in_sizes, int n_in,
                              void* d_out, int out_size, void* d_ws, size_t ws_size,
                              hipStream_t stream) {
  const float* x      = (const float*)d_in[0];
  const float* ln_g   = (const float*)d_in[1];
  const float* ln_b   = (const float*)d_in[2];
  const float* w_qkv  = (const float*)d_in[3];
  const float* w_proj = (const float*)d_in[4];
  const float* b_proj = (const float*)d_in[5];
  const float* w1     = (const float*)d_in[6];
  const float* b1     = (const float*)d_in[7];
  const float* w2     = (const float*)d_in[8];
  const float* b2     = (const float*)d_in[9];

  float* out = (float*)d_out;            // running residual stream x

  const int TOK = 8192;                  // B*N
  // per-layer transposed bf16 weight sizes (elements)
  const size_t SZ_QKV = (size_t)2304 * 768;
  const size_t SZ_PRJ = (size_t)768 * 768;
  const size_t SZ_W1  = (size_t)3072 * 768;
  const size_t SZ_W2  = (size_t)768 * 3072;
  const size_t SZ_LAYER = SZ_QKV + SZ_PRJ + SZ_W1 + SZ_W2;

  char* p = (char*)d_ws;
  unsigned short* wt = (unsigned short*)p;          p += SZ_LAYER * 6 * 2;
  float*          h  = (float*)p;                   p += (size_t)TOK * 768 * 4;
  unsigned short* hb = (unsigned short*)p;          p += (size_t)TOK * 768 * 2;
  unsigned short* qkvb = (unsigned short*)p;        p += (size_t)TOK * 2304 * 2;
  unsigned short* obb  = (unsigned short*)p;        p += (size_t)TOK * 768 * 2;
  unsigned short* mbb  = (unsigned short*)p;        p += (size_t)TOK * 3072 * 2;

  // ---- one-time per launch: transpose+convert all weights to bf16 [N][K]
  for (int i = 0; i < 6; ++i) {
    unsigned short* wl = wt + (size_t)i * SZ_LAYER;
    transpose_cvt<<<dim3(2304 / 32, 768 / 32), 256, 0, stream>>>(
        w_qkv + (size_t)i * 768 * 2304, wl, 768, 2304);
    transpose_cvt<<<dim3(768 / 32, 768 / 32), 256, 0, stream>>>(
        w_proj + (size_t)i * 768 * 768, wl + SZ_QKV, 768, 768);
    transpose_cvt<<<dim3(3072 / 32, 768 / 32), 256, 0, stream>>>(
        w1 + (size_t)i * 768 * 3072, wl + SZ_QKV + SZ_PRJ, 768, 3072);
    transpose_cvt<<<dim3(768 / 32, 3072 / 32), 256, 0, stream>>>(
        w2 + (size_t)i * 3072 * 768, wl + SZ_QKV + SZ_PRJ + SZ_W1, 3072, 768);
  }

  copy_f32<<<(TOK * 768 + 255) / 256, 256, 0, stream>>>(x, out, TOK * 768);

  for (int i = 0; i < 6; ++i) {
    const unsigned short* wl   = wt + (size_t)i * SZ_LAYER;
    const unsigned short* qkvT = wl;
    const unsigned short* prjT = wl + SZ_QKV;
    const unsigned short* w1T  = wl + SZ_QKV + SZ_PRJ;
    const unsigned short* w2T  = wl + SZ_QKV + SZ_PRJ + SZ_W1;
    const float* g   = ln_g + i * 768;
    const float* bln = ln_b + i * 768;

    // h = LN(x)   (f32 + bf16)
    ln_kernel<<<TOK, 256, 0, stream>>>(out, g, bln, h, hb);
    // qkv = h @ w_qkv   -> bf16
    gemm_bf16<<<dim3(64, 36), 256, 0, stream>>>(
        hb, qkvT, nullptr, nullptr, nullptr, qkvb, TOK, 2304, 768, 0);
    // attention -> bf16 (b, n, h*d)
    attn_kernel<<<dim3(16, 96), 256, 278528, stream>>>(qkvb, obb);
    // x = ob @ w_proj + b_proj + h   -> f32
    gemm_bf16<<<dim3(64, 12), 256, 0, stream>>>(
        obb, prjT, b_proj + i * 768, h, out, nullptr, TOK, 768, 768, 0);
    // h = LN(x)
    ln_kernel<<<TOK, 256, 0, stream>>>(out, g, bln, h, hb);
    // m = gelu(h @ w1 + b1)  -> bf16
    gemm_bf16<<<dim3(64, 48), 256, 0, stream>>>(
        hb, w1T, b1 + i * 3072, nullptr, nullptr, mbb, TOK, 3072, 768, 1);
    // x = m @ w2 + b2 + h   -> f32
    gemm_bf16<<<dim3(64, 12), 256, 0, stream>>>(
        mbb, w2T, b2 + i * 768, h, out, nullptr, TOK, 768, 3072, 0);
  }
}
